// IronRoPESelfAttention_63840393887860
// MI455X (gfx1250) — compile-verified, matrix operand order checked by reading, not run on hardware
//
#include <hip/hip_runtime.h>
#include <hip/hip_bf16.h>

// ---------------------------------------------------------------------------
// IronRoPESelfAttention for MI455X (gfx1250, wave32, WMMA + TDM)
//   B=2 T=1024 C=1024 H=16 HD=64 ROPE_M=32 FB_M=32
// Pipeline:
//   1) cvt: fp32 -> bf16 for x, W_qkv, W_proj
//   2) gemm_bf16: TDM (tensor_load_to_lds) double-buffered LDS tiles,
//      WMMA 16x16x32 bf16, s_wait_tensorcnt pipelining
//   3) rope_split -> Q,K [B,H,T,HD] bf16 and V^T [B,H,HD,T] bf16
//   4) fb_bias: bias table [T,T] fp32 (coords broadcast over B => one table)
//   5) flash_attn: per-wave 16-row q tile, online softmax, WMMA S and P@V
//   6) gemm_bf16: out = x + Y @ W_proj^T + b_proj
// ---------------------------------------------------------------------------

typedef __bf16 bf16_t;
typedef __attribute__((ext_vector_type(16))) __bf16 v16bf;
typedef __attribute__((ext_vector_type(8)))  __bf16 v8bf;
typedef __attribute__((ext_vector_type(8)))  float  v8f;
typedef unsigned int u32x4 __attribute__((ext_vector_type(4)));
typedef int          i32x4 __attribute__((ext_vector_type(4)));
typedef int          i32x8 __attribute__((ext_vector_type(8)));

enum { Bq = 2, Tq = 1024, Cq = 1024, Hq = 16, HDq = 64, RM = 32, FBM = 32 };
#define SCALE_F   0.125f            /* 1/sqrt(64)  */
#define FBSCALE_F 0.1767766953f     /* 1/sqrt(32)  */

__device__ __forceinline__ bf16_t f2bf(float f) {
  unsigned u = __builtin_bit_cast(unsigned, f);
  unsigned r = (u + 0x7fffu + ((u >> 16) & 1u)) >> 16;
  return __builtin_bit_cast(bf16_t, (unsigned short)r);
}

__device__ __forceinline__ v8f wmma_bf16(v16bf a, v16bf b, v8f c) {
  return __builtin_amdgcn_wmma_f32_16x16x32_bf16(false, a, false, b,
                                                 (short)0, c, false, false);
}

// 16x32 bf16 fragment loader (A layout; also valid for our [N,K] row-major
// "B" operands).  Lanes 0-15: row=lane, elems 0..7 = K k0..k0+7,
// elems 8..15 = K k0+16..k0+23; lanes 16-31 shift K by 8. (ISA 7.12.2)
__device__ __forceinline__ v16bf frag_load(const bf16_t* base, int ld,
                                           int row0, int k0, int lane) {
  int r  = row0 + (lane & 15);
  int kh = (lane >> 4) << 3;
  const bf16_t* p = base + (size_t)r * ld + k0 + kh;
  v8bf lo = *(const v8bf*)(p);
  v8bf hi = *(const v8bf*)(p + 16);
  return __builtin_shufflevector(lo, hi, 0,1,2,3,4,5,6,7,
                                         8,9,10,11,12,13,14,15);
}

__device__ __forceinline__ float half16_max(float v) {
  v = fmaxf(v, __shfl_xor(v, 1, 32));
  v = fmaxf(v, __shfl_xor(v, 2, 32));
  v = fmaxf(v, __shfl_xor(v, 4, 32));
  v = fmaxf(v, __shfl_xor(v, 8, 32));
  return v;
}
__device__ __forceinline__ float half16_sum(float v) {
  v += __shfl_xor(v, 1, 32);
  v += __shfl_xor(v, 2, 32);
  v += __shfl_xor(v, 4, 32);
  v += __shfl_xor(v, 8, 32);
  return v;
}

// ---- Tensor Data Mover: 2D tile (bf16) global -> LDS ----------------------
// Builds D# per ISA 08_async_tensor §8: group0 = {count=1, lds_addr,
// global_addr(57b), type=2}; group1 = {data_size=2B, tensor dims, tile dims,
// dim0 stride}; groups 2/3 zero (2D tensor).
__device__ __forceinline__ void tdm_load_2d(const bf16_t* gptr, unsigned lds_off,
                                            int dim0, int dim1,
                                            int tile0, int tile1, int stride0) {
  unsigned long long ga = (unsigned long long)(size_t)gptr;
  u32x4 g0;
  g0.x = 1u;                                          // count=1 (valid)
  g0.y = lds_off;                                     // lds_addr (bytes)
  g0.z = (unsigned)(ga & 0xffffffffu);                // global_addr[31:0]
  g0.w = (unsigned)((ga >> 32) & 0x01ffffffu) | 0x80000000u;  // addr + type=2
  i32x8 g1;
  g1[0] = 1 << 16;                                    // data_size = 2 bytes
  g1[1] = (dim0 & 0xffff) << 16;                      // tensor_dim0 lo16
  g1[2] = ((unsigned)dim0 >> 16) | ((dim1 & 0xffff) << 16);
  g1[3] = ((unsigned)dim1 >> 16) | ((tile0 & 0xffff) << 16);
  g1[4] = tile1 & 0xffff;                             // tile_dim1, tile_dim2=0
  g1[5] = stride0;                                    // dim0_stride lo32
  g1[6] = 0;
  g1[7] = 0;
  i32x4 z4 = {0, 0, 0, 0};
#if defined(__clang_major__) && (__clang_major__ >= 23)
  i32x8 z8 = {0, 0, 0, 0, 0, 0, 0, 0};
  __builtin_amdgcn_tensor_load_to_lds(g0, g1, z4, z4, z8, 0);
#else
  __builtin_amdgcn_tensor_load_to_lds(g0, g1, z4, z4, 0);
#endif
}

// ---------------------------------------------------------------- cvt fp32->bf16
__global__ void k_cvt_bf16(const float* __restrict__ in, bf16_t* __restrict__ out,
                           int n) {
  int i = blockIdx.x * blockDim.x + threadIdx.x;
  if (i < n) out[i] = f2bf(in[i]);
}

// -------------------------------------- WMMA GEMM with TDM double buffering
// out = A @ W^T + bias (+resid).  A:[M,K] bf16 rm, W:[N,K] bf16 rm.
// Block 256 thr = 8 waves, tile 128x64; K staged 64 at a time via TDM.
__global__ void k_gemm_bf16(const bf16_t* __restrict__ A,
                            const bf16_t* __restrict__ W,
                            const float* __restrict__ bias,
                            const float* __restrict__ resid,
                            float* __restrict__ out,
                            int M, int N, int K) {
  __shared__ bf16_t At[2][128][64];   // 32 KB
  __shared__ bf16_t Wt[2][64][64];    // 16 KB
  int lane = threadIdx.x & 31;
  int wave = threadIdx.x >> 5;
  int bm0 = blockIdx.x * 128;
  int n0  = blockIdx.y * 64;

  v8f acc[4];
#pragma unroll
  for (int n = 0; n < 4; ++n) acc[n] = (v8f){0,0,0,0,0,0,0,0};

  // stage 0 (wave 0 issues; TDM is a whole-wave op -> issues once)
  if (threadIdx.x < 32) {
    tdm_load_2d(A + (size_t)bm0 * K, (unsigned)(size_t)&At[0][0][0],
                K, M, 64, 128, K);
    tdm_load_2d(W + (size_t)n0 * K, (unsigned)(size_t)&Wt[0][0][0],
                K, N, 64, 64, K);
  }

  int buf = 0;
  for (int kt = 0; kt < K; kt += 64, buf ^= 1) {
    if (kt + 64 < K) {
      if (threadIdx.x < 32) {   // prefetch next stage into the other buffer
        tdm_load_2d(A + (size_t)bm0 * K + kt + 64,
                    (unsigned)(size_t)&At[buf ^ 1][0][0], K, M, 64, 128, K);
        tdm_load_2d(W + (size_t)n0 * K + kt + 64,
                    (unsigned)(size_t)&Wt[buf ^ 1][0][0], K, N, 64, 64, K);
      }
      __builtin_amdgcn_s_wait_tensorcnt(2);   // current stage's 2 DMAs done
    } else {
      __builtin_amdgcn_s_wait_tensorcnt(0);
    }
    __syncthreads();

#pragma unroll
    for (int ks = 0; ks < 2; ++ks) {
      v16bf a = frag_load(&At[buf][0][0], 64, wave * 16, ks * 32, lane);
#pragma unroll
      for (int n = 0; n < 4; ++n) {
        v16bf b = frag_load(&Wt[buf][0][0], 64, n * 16, ks * 32, lane);
        acc[n] = wmma_bf16(a, b, acc[n]);
      }
    }
    __syncthreads();   // done reading buf before it is refilled
  }

  int half = lane >> 4, col = lane & 15;
  int m0 = bm0 + wave * 16;
#pragma unroll
  for (int n = 0; n < 4; ++n) {
    int cn = n0 + n * 16 + col;
    float bv = bias[cn];
#pragma unroll
    for (int r = 0; r < 8; ++r) {
      int row = m0 + r + half * 8;
      float v = acc[n][r] + bv;
      if (resid) v += resid[(size_t)row * N + cn];
      out[(size_t)row * N + cn] = v;
    }
  }
}

// -------------------------------------------------- RoPE rotate + head split
__global__ void k_rope_split(const float* __restrict__ qkv,
                             const float* __restrict__ coords,
                             const float* __restrict__ Wr,
                             bf16_t* __restrict__ Qb, bf16_t* __restrict__ Kb,
                             bf16_t* __restrict__ Vt) {
  int bt = blockIdx.x;
  int b = bt / Tq, t = bt % Tq;
  __shared__ float cs[RM], sn[RM];
  if (threadIdx.x < RM) {
    float th = coords[t] * Wr[threadIdx.x];
    cs[threadIdx.x] = __cosf(th);
    sn[threadIdx.x] = __sinf(th);
  }
  __syncthreads();
  const float* base = qkv + (size_t)bt * (3 * Cq);
  for (int p = threadIdx.x; p < Cq / 2; p += blockDim.x) {
    int hh = p >> 5;
    int m  = p & 31;
    float c = cs[m], s = sn[m];
    float q0 = base[2 * p],          q1 = base[2 * p + 1];
    float k0 = base[Cq + 2 * p],     k1 = base[Cq + 2 * p + 1];
    float v0 = base[2 * Cq + 2 * p], v1 = base[2 * Cq + 2 * p + 1];
    size_t qo = ((size_t)(b * Hq + hh) * Tq + t) * HDq + 2 * m;
    Qb[qo]     = f2bf(q0 * c - q1 * s);
    Qb[qo + 1] = f2bf(q0 * s + q1 * c);
    Kb[qo]     = f2bf(k0 * c - k1 * s);
    Kb[qo + 1] = f2bf(k0 * s + k1 * c);
    size_t vo = ((size_t)(b * Hq + hh) * HDq + 2 * m) * Tq + t;
    Vt[vo]      = f2bf(v0);
    Vt[vo + Tq] = f2bf(v1);
  }
}

// -------------------------------------------------- Fourier bias table [T,T]
__global__ void k_fb_bias(const float* __restrict__ coords,
                          const float* __restrict__ Wfb,
                          const float* __restrict__ bcos,
                          const float* __restrict__ bsin,
                          float* __restrict__ bias) {
  __shared__ float w[FBM], bc[FBM], bs[FBM];
  if (threadIdx.x < FBM) {
    w[threadIdx.x]  = Wfb[threadIdx.x];
    bc[threadIdx.x] = bcos[threadIdx.x];
    bs[threadIdx.x] = bsin[threadIdx.x];
  }
  __syncthreads();
  int idx = blockIdx.x * blockDim.x + threadIdx.x;
  int qi = idx / Tq, ki = idx % Tq;
  float d = coords[qi] - coords[ki];
  float acc = 0.f;
#pragma unroll
  for (int m = 0; m < FBM; ++m) {
    float S = d * w[m];
    acc += __cosf(S) * bc[m] + __sinf(S) * bs[m];
  }
  bias[idx] = acc * FBSCALE_F;
}

// -------------------------------------------------- flash attention (fused)
__global__ void k_flash_attn(const bf16_t* __restrict__ Q,
                             const bf16_t* __restrict__ Kd,
                             const bf16_t* __restrict__ Vt,
                             const float* __restrict__ bias,
                             bf16_t* __restrict__ Y) {
  __shared__ bf16_t Pt[8][16][64];
  int bh = blockIdx.x;
  int b = bh / Hq, h = bh % Hq;
  int lane = threadIdx.x & 31, wave = threadIdx.x >> 5;
  int half = lane >> 4, col = lane & 15;
  int q0 = blockIdx.y * 128 + wave * 16;

  const bf16_t* qbase = Q  + (size_t)bh * Tq * HDq;
  const bf16_t* kbase = Kd + (size_t)bh * Tq * HDq;
  const bf16_t* vbase = Vt + (size_t)bh * HDq * Tq;

  v16bf qf0 = frag_load(qbase, HDq, q0, 0, lane);
  v16bf qf1 = frag_load(qbase, HDq, q0, 32, lane);

  v8f of[4];
  float mrow[8], lrow[8];
#pragma unroll
  for (int d = 0; d < 4; ++d) of[d] = (v8f){0,0,0,0,0,0,0,0};
#pragma unroll
  for (int r = 0; r < 8; ++r) { mrow[r] = -3.0e38f; lrow[r] = 0.f; }

  for (int kt = 0; kt < Tq; kt += 64) {
    if (kt + 64 < Tq)
      __builtin_prefetch(&bias[(size_t)(q0 + half * 8) * Tq + kt + 64], 0, 1);
    v8f sf[4];
#pragma unroll
    for (int n = 0; n < 4; ++n) {
      v8f s = (v8f){0,0,0,0,0,0,0,0};
      v16bf kf0 = frag_load(kbase, HDq, kt + n * 16, 0, lane);
      s = wmma_bf16(qf0, kf0, s);
      v16bf kf1 = frag_load(kbase, HDq, kt + n * 16, 32, lane);
      s = wmma_bf16(qf1, kf1, s);
      int kcol = kt + n * 16 + col;
#pragma unroll
      for (int r = 0; r < 8; ++r) {
        int qrow = q0 + r + half * 8;
        s[r] = s[r] * SCALE_F + bias[(size_t)qrow * Tq + kcol];
      }
      sf[n] = s;
    }
    float alpha[8], nm[8];
#pragma unroll
    for (int r = 0; r < 8; ++r) {
      float v = sf[0][r];
      v = fmaxf(v, sf[1][r]); v = fmaxf(v, sf[2][r]); v = fmaxf(v, sf[3][r]);
      v = half16_max(v);
      nm[r] = fmaxf(mrow[r], v);
      alpha[r] = __expf(mrow[r] - nm[r]);
      mrow[r] = nm[r];
    }
    float psum[8];
#pragma unroll
    for (int r = 0; r < 8; ++r) psum[r] = 0.f;
#pragma unroll
    for (int n = 0; n < 4; ++n)
#pragma unroll
      for (int r = 0; r < 8; ++r) {
        float p = __expf(sf[n][r] - nm[r]);
        sf[n][r] = p;
        psum[r] += p;
      }
#pragma unroll
    for (int r = 0; r < 8; ++r)
      lrow[r] = lrow[r] * alpha[r] + half16_sum(psum[r]);
#pragma unroll
    for (int d = 0; d < 4; ++d)
#pragma unroll
      for (int r = 0; r < 8; ++r) of[d][r] *= alpha[r];

#pragma unroll
    for (int n = 0; n < 4; ++n)
#pragma unroll
      for (int r = 0; r < 8; ++r)
        Pt[wave][r + half * 8][n * 16 + col] = f2bf(sf[n][r]);

    v16bf pf0 = frag_load((const bf16_t*)&Pt[wave][0][0], 64, 0, 0, lane);
    v16bf pf1 = frag_load((const bf16_t*)&Pt[wave][0][0], 64, 0, 32, lane);
#pragma unroll
    for (int d = 0; d < 4; ++d) {
      v16bf vf0 = frag_load(vbase, Tq, d * 16, kt, lane);
      of[d] = wmma_bf16(pf0, vf0, of[d]);
      v16bf vf1 = frag_load(vbase, Tq, d * 16, kt + 32, lane);
      of[d] = wmma_bf16(pf1, vf1, of[d]);
    }
  }

  float inv[8];
#pragma unroll
  for (int r = 0; r < 8; ++r) inv[r] = 1.f / lrow[r];
#pragma unroll
  for (int d = 0; d < 4; ++d)
#pragma unroll
    for (int r = 0; r < 8; ++r) {
      int qrow = q0 + r + half * 8;
      Y[(size_t)(b * Tq + qrow) * Cq + h * HDq + d * 16 + col] =
          f2bf(of[d][r] * inv[r]);
    }
}

// ---------------------------------------------------------------------------
extern "C" void kernel_launch(void* const* d_in, const int* in_sizes, int n_in,
                              void* d_out, int out_size, void* d_ws, size_t ws_size,
                              hipStream_t stream) {
  const float* x        = (const float*)d_in[0];
  const float* coords   = (const float*)d_in[1];
  const float* W_qkv    = (const float*)d_in[2];
  const float* b_qkv    = (const float*)d_in[3];
  const float* W_proj   = (const float*)d_in[4];
  const float* b_proj   = (const float*)d_in[5];
  const float* W_rope   = (const float*)d_in[6];
  const float* W_fb     = (const float*)d_in[7];
  const float* beta_cos = (const float*)d_in[8];
  const float* beta_sin = (const float*)d_in[9];
  float* out = (float*)d_out;

  char* wsp = (char*)d_ws;
  auto alloc = [&](size_t bytes) {
    char* p = wsp;
    wsp += (bytes + 255) & ~(size_t)255;
    return p;
  };
  const int M = Bq * Tq;
  bf16_t* xb     = (bf16_t*)alloc((size_t)M * Cq * 2);
  bf16_t* wqkvb  = (bf16_t*)alloc((size_t)3 * Cq * Cq * 2);
  bf16_t* wprojb = (bf16_t*)alloc((size_t)Cq * Cq * 2);
  float*  qkvf   = (float*) alloc((size_t)M * 3 * Cq * 4);
  bf16_t* Qb     = (bf16_t*)alloc((size_t)Bq * Hq * Tq * HDq * 2);
  bf16_t* Kb     = (bf16_t*)alloc((size_t)Bq * Hq * Tq * HDq * 2);
  bf16_t* Vt     = (bf16_t*)alloc((size_t)Bq * Hq * HDq * Tq * 2);
  float*  biasT  = (float*) alloc((size_t)Tq * Tq * 4);
  bf16_t* Yb     = (bf16_t*)alloc((size_t)M * Cq * 2);

  int nx = M * Cq, nw = 3 * Cq * Cq, np = Cq * Cq;
  k_cvt_bf16<<<(nx + 255) / 256, 256, 0, stream>>>(x, xb, nx);
  k_cvt_bf16<<<(nw + 255) / 256, 256, 0, stream>>>(W_qkv, wqkvb, nw);
  k_cvt_bf16<<<(np + 255) / 256, 256, 0, stream>>>(W_proj, wprojb, np);

  k_gemm_bf16<<<dim3(M / 128, (3 * Cq) / 64), 256, 0, stream>>>(
      xb, wqkvb, b_qkv, nullptr, qkvf, M, 3 * Cq, Cq);

  k_rope_split<<<Bq * Tq, 256, 0, stream>>>(qkvf, coords, W_rope, Qb, Kb, Vt);

  k_fb_bias<<<(Tq * Tq) / 256, 256, 0, stream>>>(coords, W_fb, beta_cos,
                                                 beta_sin, biasT);

  k_flash_attn<<<dim3(Bq * Hq, Tq / 128), 256, 0, stream>>>(Qb, Kb, Vt, biasT,
                                                            Yb);

  k_gemm_bf16<<<dim3(M / 128, Cq / 64), 256, 0, stream>>>(
      Yb, wprojb, b_proj, x, out, M, Cq, Cq);
}
